// CustomTransformer_64570538328578
// MI455X (gfx1250) — matrix operand order repaired and checked by reading, not compiled, not run on hardware
//
#include <hip/hip_runtime.h>
#include <math.h>

#define DEV __device__ __forceinline__

typedef __bf16 bf16;
typedef __attribute__((ext_vector_type(4)))  bf16  v4bf;
typedef __attribute__((ext_vector_type(8)))  bf16  v8bf;
typedef __attribute__((ext_vector_type(16))) bf16  v16bf;
typedef __attribute__((ext_vector_type(8)))  float v8f;

// ---- model constants (fixed by reference) ----
constexpr int cV = 32000, cD = 512, cH = 8, cLE = 6, cLD = 6, cDFF = 2048;
constexpr int cB = 4, cS = 512, cDK = 64;
constexpr int cBS = cB * cS;   // 2048 token rows
constexpr int cBH = cB * cH;   // 32 (b,h) pairs

#define NEGV (-1e9f)

DEV v16bf cat16(v8bf lo, v8bf hi) {
  return __builtin_shufflevector(lo, hi, 0,1,2,3,4,5,6,7,8,9,10,11,12,13,14,15);
}
DEV v8f wmma_bf16(v16bf a, v16bf b, v8f c) {
  return __builtin_amdgcn_wmma_f32_16x16x32_bf16(false, a, false, b, (short)0, c, false, false);
}

// Async global->LDS copy of 16 bytes (CDNA5 GLOBAL_LOAD_ASYNC_TO_LDS_B128,
// tracked with ASYNCcnt). lds_ptr: generic pointer to __shared__; its low 32
// bits are the LDS offset (generic->local lowering is a truncation on AMDGPU).
DEV void async_b128_to_lds(bf16* lds_ptr, const bf16* gptr) {
  unsigned lds_off = (unsigned)(unsigned long long)(const void*)lds_ptr;
  asm volatile("global_load_async_to_lds_b128 %0, %1, off"
               :: "v"(lds_off), "v"(gptr)
               : "memory");
}
DEV void wait_async0() { asm volatile("s_wait_asynccnt 0" ::: "memory"); }

// =====================================================================
// Weight convert: fp32 [K,N] row-major -> bf16 [N,K] row-major (transposed)
// grid (N/32, K/32, batch), block 256
// =====================================================================
__global__ void wtconv_kernel(const float* __restrict__ in, bf16* __restrict__ out,
                              int Kd, int N) {
  __shared__ float tile[32][33];
  const int bz = blockIdx.z;
  const float* ip = in + (size_t)bz * Kd * N;
  bf16* op = out + (size_t)bz * Kd * N;
  const int tx = threadIdx.x & 31, ty = threadIdx.x >> 5;
  const int n0 = blockIdx.x * 32, k0 = blockIdx.y * 32;
#pragma unroll
  for (int i = 0; i < 4; i++) {
    int r = ty + i * 8;
    tile[r][tx] = ip[(size_t)(k0 + r) * N + n0 + tx];
  }
  __syncthreads();
#pragma unroll
  for (int i = 0; i < 4; i++) {
    int r = ty + i * 8;
    op[(size_t)(n0 + r) * Kd + k0 + tx] = (bf16)tile[tx][r];
  }
}

// =====================================================================
// Embedding: out[row, :] = emb[tok[row], :] * scale + pe[row % S, :]
// grid (BS), block 256
// =====================================================================
__global__ void embed_kernel(const int* __restrict__ toks, const float* __restrict__ emb,
                             const float* __restrict__ pe, float* __restrict__ out,
                             float scale) {
  const int row = blockIdx.x;
  const int s = row % cS;
  const int t = toks[row];
  for (int d = threadIdx.x; d < cD; d += blockDim.x)
    out[(size_t)row * cD + d] = emb[(size_t)t * cD + d] * scale + pe[(size_t)s * cD + d];
}

// =====================================================================
// Main GEMM: C[M,N] = relu?( A[M,K](f32) * Wt[N,K](bf16)^T + bias[N] )
// block 256 threads = 8 waves; tile 128x128; each wave does 32x64 (2x4 WMMA tiles)
// A staged via VGPR (needs f32->bf16 cvt); B staged via async-to-LDS.
// __launch_bounds__(256,1): allow full VGPR budget -> no accumulator spills.
// grid (N/128, M/128)
// =====================================================================
__global__ void __launch_bounds__(256, 1)
gemm_kernel(const float* __restrict__ A, const bf16* __restrict__ Wt,
            const float* __restrict__ bias, float* __restrict__ C,
            int M, int N, int Kd, int relu) {
  __shared__ bf16 As[128 * 40];
  __shared__ bf16 Bs[128 * 40];
  const int tid = threadIdx.x;
  const int lane = tid & 31;
  const int w = tid >> 5;
  const int wm = w >> 1;          // 0..3 -> rows wm*32
  const int wn = w & 1;           // 0..1 -> cols wn*64
  const int lr = lane & 15;
  const int khalf = lane >> 4;
  const int bm = blockIdx.y * 128, bn = blockIdx.x * 128;

  v8f acc[2][4];
#pragma unroll
  for (int i = 0; i < 2; i++)
#pragma unroll
    for (int j = 0; j < 4; j++)
#pragma unroll
      for (int e = 0; e < 8; e++) acc[i][j][e] = 0.0f;

  for (int kt = 0; kt < Kd; kt += 32) {
    // stage B tile: 128 n-rows x 32 k of bf16, global -> LDS async (no VGPR hop)
#pragma unroll
    for (int g = 0; g < 2; g++) {
      int gi = tid + g * 256;            // 0..511
      int r = gi >> 2;                   // 0..127
      int c8 = (gi & 3) * 8;
      async_b128_to_lds(&Bs[r * 40 + c8], Wt + (size_t)(bn + r) * Kd + kt + c8);
    }
    // stage A tile: 128 rows x 32 k, fp32 -> bf16 (VALU convert)
#pragma unroll
    for (int g = 0; g < 4; g++) {
      int li = (tid + g * 256) * 4;      // 0..4095
      int r = li >> 5;                   // /32
      int c = li & 31;
      const float4 av = *(const float4*)(A + (size_t)(bm + r) * Kd + kt + c);
      if (g == 0 && kt + 32 < Kd)        // prefetch next A slab behind the WMMAs
        __builtin_prefetch(A + (size_t)(bm + r) * Kd + kt + 32 + c, 0, 3);
      v4bf t4;
      t4[0] = (bf16)av.x; t4[1] = (bf16)av.y; t4[2] = (bf16)av.z; t4[3] = (bf16)av.w;
      *(v4bf*)(&As[r * 40 + c]) = t4;
    }
    wait_async0();                        // B tile landed in LDS
    __syncthreads();                      // (compiler also waits DScnt for As stores)

    v16bf af[2], bfr[4];
#pragma unroll
    for (int i = 0; i < 2; i++) {
      const bf16* pa = &As[(wm * 32 + i * 16 + lr) * 40];
      v8bf lo = *(const v8bf*)(pa + khalf * 8);
      v8bf hi = *(const v8bf*)(pa + 16 + khalf * 8);
      af[i] = cat16(lo, hi);
    }
#pragma unroll
    for (int j = 0; j < 4; j++) {
      const bf16* pb = &Bs[(wn * 64 + j * 16 + lr) * 40 + khalf * 8];
      v8bf lo = *(const v8bf*)(pb);
      v8bf hi = *(const v8bf*)(pb + 16);
      bfr[j] = cat16(lo, hi);
    }
#pragma unroll
    for (int i = 0; i < 2; i++)
#pragma unroll
      for (int j = 0; j < 4; j++)
        acc[i][j] = wmma_bf16(af[i], bfr[j], acc[i][j]);
    __syncthreads();
  }

  // epilogue
#pragma unroll
  for (int i = 0; i < 2; i++) {
#pragma unroll
    for (int j = 0; j < 4; j++) {
      int c = bn + wn * 64 + j * 16 + lr;
      float bv = bias[c];
#pragma unroll
      for (int e = 0; e < 8; e++) {
        int r = bm + wm * 32 + i * 16 + e + 8 * khalf;
        float v = acc[i][j][e] + bv;
        if (relu) v = fmaxf(v, 0.0f);
        C[(size_t)r * N + c] = v;
      }
    }
  }
}

// =====================================================================
// Attention scores: S[bh,q,k] = mask ? (Q.K^T)*scale : NEG
// Q,K fp32 [B*S, D] token-major; one wave per 16x16 tile
// grid (S/16, S/16, B*H), block 32
// mode 0: key padding mask (toks = src), mode 1: causal + query padding (toks = tgt)
// =====================================================================
__global__ void scores_kernel(const float* __restrict__ Q, const float* __restrict__ Kb,
                              const int* __restrict__ toks, float* __restrict__ Sout,
                              int mode, float scale) {
  const int lane = threadIdx.x;
  const int lr = lane & 15, khalf = lane >> 4;
  const int bh = blockIdx.z;
  const int b = bh >> 3, h = bh & 7;           // H = 8
  const int qBase = blockIdx.y * 16, kBase = blockIdx.x * 16;

  const float* qp = Q + (size_t)(b * cS + qBase + lr) * cD + h * cDK;
  const float* kp = Kb + (size_t)(b * cS + kBase + lr) * cD + h * cDK;

  v8f acc;
#pragma unroll
  for (int e = 0; e < 8; e++) acc[e] = 0.0f;

#pragma unroll
  for (int kt = 0; kt < cDK; kt += 32) {
    v16bf a, bb;
    const int ka = kt + khalf * 8;
#pragma unroll
    for (int j = 0; j < 8; j++) { a[j] = (bf16)qp[ka + j]; a[j + 8] = (bf16)qp[ka + 16 + j]; }
    const int kb2 = kt + khalf * 8;
#pragma unroll
    for (int j = 0; j < 8; j++) { bb[j] = (bf16)kp[kb2 + j]; bb[j + 8] = (bf16)kp[kb2 + 16 + j]; }
    acc = wmma_bf16(a, bb, acc);
  }

#pragma unroll
  for (int e = 0; e < 8; e++) {
    int q = qBase + e + 8 * khalf;
    int tk = kBase + lr;
    bool ok = (mode == 0) ? (toks[b * cS + tk] != 0)
                          : ((toks[b * cS + q] != 0) && (tk <= q));
    Sout[(size_t)(bh * cS + q) * cS + tk] = ok ? acc[e] * scale : NEGV;
  }
}

// =====================================================================
// Softmax over rows of scores -> bf16 probs. grid (B*H*S), block 256
// =====================================================================
__global__ void softmax_kernel(const float* __restrict__ Sc, bf16* __restrict__ P) {
  __shared__ float sh[256];
  const size_t base = (size_t)blockIdx.x * cS;
  const int t = threadIdx.x;
  float v0 = Sc[base + t], v1 = Sc[base + t + 256];
  sh[t] = fmaxf(v0, v1);
  __syncthreads();
  for (int s = 128; s > 0; s >>= 1) { if (t < s) sh[t] = fmaxf(sh[t], sh[t + s]); __syncthreads(); }
  float m = sh[0];
  __syncthreads();
  float e0 = __expf(v0 - m), e1 = __expf(v1 - m);
  sh[t] = e0 + e1;
  __syncthreads();
  for (int s = 128; s > 0; s >>= 1) { if (t < s) sh[t] += sh[t + s]; __syncthreads(); }
  float inv = 1.0f / sh[0];
  P[base + t] = (bf16)(e0 * inv);
  P[base + t + 256] = (bf16)(e1 * inv);
}

// =====================================================================
// V transpose: fp32 [B*S, D] -> bf16 Vt [B*H, DK, S]. grid (B*H*DK), block 256
// =====================================================================
__global__ void vtrans_kernel(const float* __restrict__ Vb, bf16* __restrict__ Vt) {
  const int r = blockIdx.x;            // 0..BH*DK-1
  const int bh = r / cDK, d = r % cDK;
  const int b = bh >> 3, h = bh & 7;
  for (int s = threadIdx.x; s < cS; s += blockDim.x)
    Vt[(size_t)r * cS + s] = (bf16)Vb[(size_t)(b * cS + s) * cD + h * cDK + d];
}

// =====================================================================
// Context: ctx[b,q,h*DK+d] = sum_k P[bh,q,k] * V[b,k,h,d]
// P bf16 [BH,S,S], Vt bf16 [BH,DK,S]; one wave per 16x16 tile
// grid (DK/16, S/16, B*H), block 32
// =====================================================================
__global__ void ctx_kernel(const bf16* __restrict__ P, const bf16* __restrict__ Vt,
                           float* __restrict__ ctx) {
  const int lane = threadIdx.x;
  const int lr = lane & 15, khalf = lane >> 4;
  const int bh = blockIdx.z;
  const int b = bh >> 3, h = bh & 7;
  const int qBase = blockIdx.y * 16, dBase = blockIdx.x * 16;

  const bf16* pp = P + (size_t)(bh * cS + qBase + lr) * cS;
  const bf16* vp = Vt + (size_t)(bh * cDK + dBase + lr) * cS;

  v8f acc;
#pragma unroll
  for (int e = 0; e < 8; e++) acc[e] = 0.0f;

  for (int kt = 0; kt < cS; kt += 32) {
    v8bf alo = *(const v8bf*)(pp + kt + khalf * 8);
    v8bf ahi = *(const v8bf*)(pp + kt + 16 + khalf * 8);
    v8bf blo = *(const v8bf*)(vp + kt + khalf * 8);
    v8bf bhi = *(const v8bf*)(vp + kt + 16 + khalf * 8);
    acc = wmma_bf16(cat16(alo, ahi), cat16(blo, bhi), acc);
  }

#pragma unroll
  for (int e = 0; e < 8; e++) {
    int q = qBase + e + 8 * khalf;
    ctx[(size_t)(b * cS + q) * cD + h * cDK + dBase + lr] = acc[e];
  }
}

// =====================================================================
// LayerNorm(residual): out[row,:] = g*( (x+a) - mu )*rsqrt(var+eps) + b
// grid (BS), block 256, D=512 (2 elems per thread). out may alias x.
// =====================================================================
__global__ void ln_kernel(const float* __restrict__ x, const float* __restrict__ a,
                          const float* __restrict__ g, const float* __restrict__ bta,
                          float* __restrict__ out) {
  __shared__ float sh[256];
  const size_t base = (size_t)blockIdx.x * cD;
  const int t = threadIdx.x;
  float v0 = x[base + t] + a[base + t];
  float v1 = x[base + t + 256] + a[base + t + 256];
  sh[t] = v0 + v1;
  __syncthreads();
  for (int s = 128; s > 0; s >>= 1) { if (t < s) sh[t] += sh[t + s]; __syncthreads(); }
  float mu = sh[0] * (1.0f / cD);
  __syncthreads();
  float d0 = v0 - mu, d1 = v1 - mu;
  sh[t] = d0 * d0 + d1 * d1;
  __syncthreads();
  for (int s = 128; s > 0; s >>= 1) { if (t < s) sh[t] += sh[t + s]; __syncthreads(); }
  float rstd = rsqrtf(sh[0] * (1.0f / cD) + 1e-5f);
  out[base + t] = g[t] * d0 * rstd + bta[t];
  out[base + t + 256] = g[t + 256] * d1 * rstd + bta[t + 256];
}

// =====================================================================
// Host orchestration
// =====================================================================
extern "C" void kernel_launch(void* const* d_in, const int* in_sizes, int n_in,
                              void* d_out, int out_size, void* d_ws, size_t ws_size,
                              hipStream_t stream) {
  (void)in_sizes; (void)n_in; (void)out_size; (void)ws_size;

  const int*   src        = (const int*)d_in[0];
  const int*   tgt        = (const int*)d_in[1];
  const float* enc_emb    = (const float*)d_in[2];
  const float* dec_emb    = (const float*)d_in[3];
  const float* pe         = (const float*)d_in[4];
  const float* enc_attn_w = (const float*)d_in[5];
  const float* enc_attn_b = (const float*)d_in[6];
  const float* enc_ffn_w1 = (const float*)d_in[7];
  const float* enc_ffn_b1 = (const float*)d_in[8];
  const float* enc_ffn_w2 = (const float*)d_in[9];
  const float* enc_ffn_b2 = (const float*)d_in[10];
  const float* enc_ln_g   = (const float*)d_in[11];
  const float* enc_ln_bb  = (const float*)d_in[12];
  const float* dec_sa_w   = (const float*)d_in[13];
  const float* dec_sa_b   = (const float*)d_in[14];
  const float* dec_ca_w   = (const float*)d_in[15];
  const float* dec_ca_b   = (const float*)d_in[16];
  const float* dec_ffn_w1 = (const float*)d_in[17];
  const float* dec_ffn_b1 = (const float*)d_in[18];
  const float* dec_ffn_w2 = (const float*)d_in[19];
  const float* dec_ffn_b2 = (const float*)d_in[20];
  const float* dec_ln_g   = (const float*)d_in[21];
  const float* dec_ln_bb  = (const float*)d_in[22];
  const float* out_w      = (const float*)d_in[23];
  const float* out_b      = (const float*)d_in[24];
  float* out = (float*)d_out;

  // ---- carve workspace ----
  char* p = (char*)d_ws;
  auto carve = [&](size_t elems, size_t esz) -> void* {
    void* r = (void*)p;
    p += (elems * esz + 255) & ~(size_t)255;
    return r;
  };
  bf16* wt_ea  = (bf16*)carve((size_t)cLE * 4 * cD * cD, 2);
  bf16* wt_ef1 = (bf16*)carve((size_t)cLE * cD * cDFF, 2);
  bf16* wt_ef2 = (bf16*)carve((size_t)cLE * cDFF * cD, 2);
  bf16* wt_sa  = (bf16*)carve((size_t)cLD * 4 * cD * cD, 2);
  bf16* wt_ca  = (bf16*)carve((size_t)cLD * 4 * cD * cD, 2);
  bf16* wt_df1 = (bf16*)carve((size_t)cLD * cD * cDFF, 2);
  bf16* wt_df2 = (bf16*)carve((size_t)cLD * cDFF * cD, 2);
  bf16* wt_out = (bf16*)carve((size_t)cD * cV, 2);
  float* xb = (float*)carve((size_t)cBS * cD, 4);
  float* yb = (float*)carve((size_t)cBS * cD, 4);
  float* qb = (float*)carve((size_t)cBS * cD, 4);
  float* kb = (float*)carve((size_t)cBS * cD, 4);
  float* vb = (float*)carve((size_t)cBS * cD, 4);
  float* t1 = (float*)carve((size_t)cBS * cDFF, 4);
  float* t2 = (float*)carve((size_t)cBS * cD, 4);
  float* sc = (float*)carve((size_t)cBH * cS * cS, 4);
  bf16*  pb = (bf16*)carve((size_t)cBH * cS * cS, 2);
  bf16*  vt = (bf16*)carve((size_t)cBH * cDK * cS, 2);

  // ---- convert all weights to transposed bf16 ----
  wtconv_kernel<<<dim3(cD / 32, cD / 32, cLE * 4), 256, 0, stream>>>(enc_attn_w, wt_ea, cD, cD);
  wtconv_kernel<<<dim3(cDFF / 32, cD / 32, cLE), 256, 0, stream>>>(enc_ffn_w1, wt_ef1, cD, cDFF);
  wtconv_kernel<<<dim3(cD / 32, cDFF / 32, cLE), 256, 0, stream>>>(enc_ffn_w2, wt_ef2, cDFF, cD);
  wtconv_kernel<<<dim3(cD / 32, cD / 32, cLD * 4), 256, 0, stream>>>(dec_sa_w, wt_sa, cD, cD);
  wtconv_kernel<<<dim3(cD / 32, cD / 32, cLD * 4), 256, 0, stream>>>(dec_ca_w, wt_ca, cD, cD);
  wtconv_kernel<<<dim3(cDFF / 32, cD / 32, cLD), 256, 0, stream>>>(dec_ffn_w1, wt_df1, cD, cDFF);
  wtconv_kernel<<<dim3(cD / 32, cDFF / 32, cLD), 256, 0, stream>>>(dec_ffn_w2, wt_df2, cDFF, cD);
  wtconv_kernel<<<dim3(cV / 32, cD / 32, 1), 256, 0, stream>>>(out_w, wt_out, cD, cV);

  auto gemm = [&](const float* A, const bf16* Wt, const float* bias, float* C,
                  int M, int N, int Kd, int relu) {
    gemm_kernel<<<dim3(N / 128, M / 128), 256, 0, stream>>>(A, Wt, bias, C, M, N, Kd, relu);
  };

  // multi-head attention: result (pre-residual) lands in qb
  auto mha = [&](const float* xq, const float* xkv, const bf16* w4, const float* b4,
                 const int* toks, int mode) {
    gemm(xq,  w4 + 0 * (size_t)cD * cD, b4 + 0 * cD, qb, cBS, cD, cD, 0);
    gemm(xkv, w4 + 1 * (size_t)cD * cD, b4 + 1 * cD, kb, cBS, cD, cD, 0);
    gemm(xkv, w4 + 2 * (size_t)cD * cD, b4 + 2 * cD, vb, cBS, cD, cD, 0);
    vtrans_kernel<<<dim3(cBH * cDK), 256, 0, stream>>>(vb, vt);
    scores_kernel<<<dim3(cS / 16, cS / 16, cBH), 32, 0, stream>>>(qb, kb, toks, sc, mode, 0.125f);
    softmax_kernel<<<dim3(cBH * cS), 256, 0, stream>>>(sc, pb);
    ctx_kernel<<<dim3(cDK / 16, cS / 16, cBH), 32, 0, stream>>>(pb, vt, t2);
    gemm(t2, w4 + 3 * (size_t)cD * cD, b4 + 3 * cD, qb, cBS, cD, cD, 0);
  };

  const float embScale = sqrtf((float)cD);

  // ---- encoder ----
  embed_kernel<<<dim3(cBS), 256, 0, stream>>>(src, enc_emb, pe, xb, embScale);
  for (int i = 0; i < cLE; i++) {
    mha(xb, xb, wt_ea + (size_t)i * 4 * cD * cD, enc_attn_b + (size_t)i * 4 * cD, src, 0);
    ln_kernel<<<dim3(cBS), 256, 0, stream>>>(xb, qb,
        enc_ln_g + (size_t)(i * 2 + 0) * cD, enc_ln_bb + (size_t)(i * 2 + 0) * cD, xb);
    gemm(xb, wt_ef1 + (size_t)i * cD * cDFF, enc_ffn_b1 + (size_t)i * cDFF, t1, cBS, cDFF, cD, 1);
    gemm(t1, wt_ef2 + (size_t)i * cDFF * cD, enc_ffn_b2 + (size_t)i * cD, t2, cBS, cD, cDFF, 0);
    ln_kernel<<<dim3(cBS), 256, 0, stream>>>(xb, t2,
        enc_ln_g + (size_t)(i * 2 + 1) * cD, enc_ln_bb + (size_t)(i * 2 + 1) * cD, xb);
  }
  // xb now holds enc_out (untouched below except as K/V source)

  // ---- decoder ----
  embed_kernel<<<dim3(cBS), 256, 0, stream>>>(tgt, dec_emb, pe, yb, embScale);
  for (int i = 0; i < cLD; i++) {
    mha(yb, yb, wt_sa + (size_t)i * 4 * cD * cD, dec_sa_b + (size_t)i * 4 * cD, tgt, 1);
    ln_kernel<<<dim3(cBS), 256, 0, stream>>>(yb, qb,
        dec_ln_g + (size_t)(i * 3 + 0) * cD, dec_ln_bb + (size_t)(i * 3 + 0) * cD, yb);
    mha(yb, xb, wt_ca + (size_t)i * 4 * cD * cD, dec_ca_b + (size_t)i * 4 * cD, src, 0);
    ln_kernel<<<dim3(cBS), 256, 0, stream>>>(yb, qb,
        dec_ln_g + (size_t)(i * 3 + 1) * cD, dec_ln_bb + (size_t)(i * 3 + 1) * cD, yb);
    gemm(yb, wt_df1 + (size_t)i * cD * cDFF, dec_ffn_b1 + (size_t)i * cDFF, t1, cBS, cDFF, cD, 1);
    gemm(t1, wt_df2 + (size_t)i * cDFF * cD, dec_ffn_b2 + (size_t)i * cD, t2, cBS, cD, cDFF, 0);
    ln_kernel<<<dim3(cBS), 256, 0, stream>>>(yb, t2,
        dec_ln_g + (size_t)(i * 3 + 2) * cD, dec_ln_bb + (size_t)(i * 3 + 2) * cD, yb);
  }

  // ---- vocab projection ----
  gemm(yb, wt_out, out_b, out, cBS, cV, cD, 0);
}
